// LSTMLayer_41403484733559
// MI455X (gfx1250) — matrix operand order repaired and checked by reading, not compiled, not run on hardware
//
#include <hip/hip_runtime.h>
#include <hip/hip_bf16.h>
#include <math.h>

// ---------------------------------------------------------------------------
// LSTM cell + vocab projection + log_softmax for MI455X (gfx1250, wave32).
// - GEMMs on v_wmma_f32_16x16x32_bf16, fp32 accumulation.
// - Activations packed to bf16 once in ws; A-tiles staged into LDS via the
//   Tensor Data Mover (tensor_load_to_lds + s_wait_tensorcnt); W-tiles
//   converted fp32->bf16 in-register (v_add + v_perm_b32 pack) from
//   non-temporal loads so the streamed W_out (412 MB) does not evict the
//   L2-resident A matrix (16 MB).
// - Grid order: M-tiles fastest so concurrent blocks share one W panel ->
//   W_out streams from HBM exactly once (~1.3 GB total HBM ~= 55 us floor).
// ---------------------------------------------------------------------------

#define DIM_INPUT  1024
#define DIM_HIDDEN 1024
#define DIM_K      2048      // dim_input + dim_hidden
#define NGATE      4096      // 4 stacked gates x 1024
#define BATCH      4096
#define VOCAB      50257

typedef __attribute__((ext_vector_type(16))) __bf16 v16bf;
typedef __attribute__((ext_vector_type(8)))  __bf16 v8bf;
typedef __attribute__((ext_vector_type(8)))  float  v8f;
typedef __attribute__((ext_vector_type(4)))  float  v4f;
typedef __attribute__((ext_vector_type(4)))  unsigned int v4u;
typedef __attribute__((ext_vector_type(4)))  int    v4i;
typedef __attribute__((ext_vector_type(8)))  int    v8i;

__device__ __forceinline__ __bf16 f2bf(float f) {
  // round-to-nearest-even fp32 -> bf16 (used on low-volume paths)
  unsigned u = __builtin_bit_cast(unsigned, f);
  unsigned r = u + 0x7FFFu + ((u >> 16) & 1u);
  unsigned short h = (unsigned short)(r >> 16);
  return __builtin_bit_cast(__bf16, h);
}

// Pack two fp32 into two bf16 (round-half-up) in one dword: 2x v_add + v_perm.
__device__ __forceinline__ unsigned pack2bf(float f0, float f1) {
  unsigned u0 = __builtin_bit_cast(unsigned, f0) + 0x8000u;
  unsigned u1 = __builtin_bit_cast(unsigned, f1) + 0x8000u;
  // result bytes: [0]=u0.b2 [1]=u0.b3 [2]=u1.b2 [3]=u1.b3
  return __builtin_amdgcn_perm(u1, u0, 0x07060302u);
}

__device__ __forceinline__ float sigmoid_f(float x) {
  return 1.0f / (1.0f + __expf(-x));
}

// ---------------------------------------------------------------------------
// TDM: load a [rows=128, cols=32] bf16 tile from a row-major [4096 x 2048]
// bf16 tensor into LDS. D# per CDNA5 ISA.
// gaddr points at the tile start; LDS receives 128 contiguous 64B rows.
// ---------------------------------------------------------------------------
__device__ __forceinline__ void tdm_load_a_tile(const __bf16* gaddr, unsigned lds_addr) {
  unsigned long long ga = (unsigned long long)(uintptr_t)gaddr;
  v4u g0;
  g0[0] = 1u;                                                // count=1 (valid), user mode
  g0[1] = lds_addr;                                          // lds_addr [63:32]
  g0[2] = (unsigned)(ga & 0xFFFFFFFFu);                      // global_addr[31:0]
  g0[3] = (unsigned)((ga >> 32) & 0x01FFFFFFu) | (2u << 30); // global_addr[56:32] | type=2
  v8i g1;
  g1[0] = 1 << 16;                    // workgroup_mask=0, data_size=1 (2 bytes)
  g1[1] = (int)((unsigned)(DIM_K & 0xFFFF) << 16);   // tensor_dim0[15:0] @ bits 63:48
  g1[2] = (int)(((unsigned)DIM_K >> 16) |            // tensor_dim0[31:16]
                ((unsigned)(BATCH & 0xFFFF) << 16)); // tensor_dim1[15:0]
  g1[3] = (int)(((unsigned)BATCH >> 16) |            // tensor_dim1[31:16]
                (32u << 16));                        // tile_dim0 = 32
  g1[4] = 128;                        // tile_dim1 = 128, tile_dim2 = 0
  g1[5] = DIM_K;                      // tensor_dim0_stride[31:0] = 2048
  g1[6] = 0;                          // stride msbs / tensor_dim1_stride lo
  g1[7] = 0;
  v4i z4 = {};
#if __clang_major__ >= 23
  v8i z8 = {};
  __builtin_amdgcn_tensor_load_to_lds(g0, g1, z4, z4, z8, 0);
#else
  __builtin_amdgcn_tensor_load_to_lds(g0, g1, z4, z4, 0);
#endif
}

// ---------------------------------------------------------------------------
// Pack input|H into bf16 activation matrix Abf [4096][2048].
// ---------------------------------------------------------------------------
__global__ __launch_bounds__(256)
void pack_a_kernel(const float* __restrict__ X, const float* __restrict__ H,
                   __bf16* __restrict__ Abf) {
  const int idx = blockIdx.x * blockDim.x + threadIdx.x;    // over 4096*1024
  const int b = idx >> 10, j = idx & 1023;
  Abf[(size_t)b * DIM_K + j]              = f2bf(X[idx]);
  Abf[(size_t)b * DIM_K + DIM_INPUT + j]  = f2bf(H[idx]);
}

// ---------------------------------------------------------------------------
// Tiled WMMA GEMM:  Out[m][n] = epilogue( Abf[m] . W[n] + b[n] )
// MODE 0: gate GEMM  (W = 4 stacked gate weights, activation in epilogue)
// MODE 1: vocab GEMM (W = W_out, bias only, n bounds-checked, NT stores)
// Macro tile 128x128, 8 waves (2 M x 4 N), wave tile 64x32, K-step 32.
// Grid: blockIdx.x = M tile (fastest -> W panel shared), blockIdx.y = N tile.
// ---------------------------------------------------------------------------
template <int MODE>
__global__ __launch_bounds__(256)
void wmma_gemm_kernel(const __bf16* __restrict__ Abf,
                      const float* __restrict__ W0, const float* __restrict__ W1,
                      const float* __restrict__ W2, const float* __restrict__ W3,
                      const float* __restrict__ b0, const float* __restrict__ b1,
                      const float* __restrict__ b2, const float* __restrict__ b3,
                      float* __restrict__ Out, int ldOut, int nTotal) {
  const int tid   = threadIdx.x;
  const int lane  = tid & 31;
  const int wv    = tid >> 5;
  const int half  = lane >> 4;
  const int ln    = lane & 15;
  const int waveM = wv & 1;      // 2 waves along M
  const int waveN = wv >> 1;     // 4 waves along N
  const int mBlock = blockIdx.x * 128;   // M fastest
  const int nBlock = blockIdx.y * 128;

  struct SH {
    __bf16 As[2][128][32];
    __bf16 Bs[2][128][32];
  };
  __shared__ alignas(64) SH sh;
  const unsigned ldsA0 = (unsigned)(uintptr_t)(&sh.As[0][0][0]);
  const unsigned ldsA1 = (unsigned)(uintptr_t)(&sh.As[1][0][0]);

  // ---- B loader: each thread owns a 16-float chunk of a [128 x 32] W tile
  const int ldRow = tid >> 1;
  const int ldSeg = (tid & 1) * 16;

  int gate = 0;
  const float* wBase;
  int wRow;
  if (MODE == 0) {
    gate  = nBlock >> 10;                           // tile fully inside one gate
    wBase = (gate == 0) ? W0 : (gate == 1) ? W1 : (gate == 2) ? W2 : W3;
    wRow  = (nBlock & 1023) + ldRow;
  } else {
    wBase = W0;
    wRow  = nBlock + ldRow;
    if (wRow > nTotal - 1) wRow = nTotal - 1;       // clamp; store is guarded
  }
  const float* wThread = wBase + (size_t)wRow * DIM_K + ldSeg;
  const __bf16* aTileBase = Abf + (size_t)mBlock * DIM_K;

  v8f acc[4][2] = {};
  v4f rb[4];

  auto loadB = [&](int kt) {
    const v4f* p4 = (const v4f*)(wThread + kt * 32);
    rb[0] = __builtin_nontemporal_load(p4 + 0);
    rb[1] = __builtin_nontemporal_load(p4 + 1);
    rb[2] = __builtin_nontemporal_load(p4 + 2);
    rb[3] = __builtin_nontemporal_load(p4 + 3);
  };
  auto stageB = [&](int buf) {
    unsigned* db = (unsigned*)&sh.Bs[buf][ldRow][ldSeg];
    const float* fb = (const float*)rb;
    uint4 w0, w1;
    w0.x = pack2bf(fb[0],  fb[1]);   w0.y = pack2bf(fb[2],  fb[3]);
    w0.z = pack2bf(fb[4],  fb[5]);   w0.w = pack2bf(fb[6],  fb[7]);
    w1.x = pack2bf(fb[8],  fb[9]);   w1.y = pack2bf(fb[10], fb[11]);
    w1.z = pack2bf(fb[12], fb[13]);  w1.w = pack2bf(fb[14], fb[15]);
    *(uint4*)(db)     = w0;
    *(uint4*)(db + 4) = w1;
  };

  auto compute = [&](int buf) {
    v16bf afr[4], bfr[2];
#pragma unroll
    for (int mi = 0; mi < 4; ++mi) {
      // 16-bit A layout (contiguous, matches B): lanes 0-15 hold K {0..15},
      // lanes 16-31 hold K {16..31} for row m = lane&15.
      const int row = waveM * 64 + mi * 16 + ln;
      v8bf lo = *(const v8bf*)&sh.As[buf][row][16 * half];
      v8bf hi = *(const v8bf*)&sh.As[buf][row][16 * half + 8];
      afr[mi] = __builtin_shufflevector(lo, hi, 0,1,2,3,4,5,6,7,8,9,10,11,12,13,14,15);
    }
#pragma unroll
    for (int ni = 0; ni < 2; ++ni) {
      const int col = waveN * 32 + ni * 16 + ln;
      bfr[ni] = *(const v16bf*)&sh.Bs[buf][col][16 * half];
    }
#pragma unroll
    for (int mi = 0; mi < 4; ++mi)
#pragma unroll
      for (int ni = 0; ni < 2; ++ni)
        acc[mi][ni] = __builtin_amdgcn_wmma_f32_16x16x32_bf16(
            false, afr[mi], false, bfr[ni], (short)0, acc[mi][ni], false, false);
  };

  // ---- K loop: TDM double-buffers A tiles, threads double-buffer B tiles
  const int KT = DIM_K / 32;   // 64 iterations
  if (wv == 0) tdm_load_a_tile(aTileBase + 0, ldsA0);
  loadB(0);
  stageB(0);
  int buf = 0;
  for (int kt = 0; kt < KT; ++kt) {
    if (wv == 0) __builtin_amdgcn_s_wait_tensorcnt(0);  // A tile for `buf` landed
    __syncthreads();                                    // + B staging visible
    const bool more = (kt + 1) < KT;
    if (more) {
      if (wv == 0)
        tdm_load_a_tile(aTileBase + (kt + 1) * 32, buf ? ldsA0 : ldsA1);
      loadB(kt + 1);
      if (kt + 2 < KT) __builtin_prefetch(wThread + (kt + 2) * 32, 0, 1);
    }
    compute(buf);
    if (more) stageB(buf ^ 1);
    buf ^= 1;
  }

  // ---- epilogue: bias (+ activation for gates), store fp32
  const int nColBase = nBlock + waveN * 32;
#pragma unroll
  for (int ni = 0; ni < 2; ++ni) {
    const int n = nColBase + ni * 16 + ln;
    float bias;
    if (MODE == 0) {
      const int r = n & 1023;
      bias = ((gate == 0) ? b0 : (gate == 1) ? b1 : (gate == 2) ? b2 : b3)[r];
    } else {
      if (n >= nTotal) continue;     // guard partial last N-tile (50257)
      bias = b0[n];
    }
#pragma unroll
    for (int mi = 0; mi < 4; ++mi) {
      const int mTile = mBlock + waveM * 64 + mi * 16 + 8 * half;
#pragma unroll
      for (int r = 0; r < 8; ++r) {
        float v = acc[mi][ni][r] + bias;
        if (MODE == 0) {
          v = (gate == 2) ? tanhf(v) : sigmoid_f(v);
          Out[(size_t)(mTile + r) * ldOut + n] = v;
        } else {
          __builtin_nontemporal_store(v, &Out[(size_t)(mTile + r) * ldOut + n]);
        }
      }
    }
  }
}

// ---------------------------------------------------------------------------
// Pointwise LSTM update; also refreshes the bf16 H half of the packed A.
// G layout: [b][4096] with f=0..1023, i=1024.., c=2048.., o=3072.. (activated)
// ---------------------------------------------------------------------------
__global__ __launch_bounds__(256)
void lstm_pointwise(const float* __restrict__ G, const float* __restrict__ C,
                    float* __restrict__ Hnew, float* __restrict__ Cnew,
                    __bf16* __restrict__ Abf) {
  const int idx = blockIdx.x * blockDim.x + threadIdx.x;   // over 4096*1024
  const int b = idx >> 10, j = idx & 1023;
  const float* g = G + (size_t)b * NGATE;
  const float f = g[j];
  const float i = g[1024 + j];
  const float c = g[2048 + j];
  const float o = g[3072 + j];
  const float cn = C[idx] * f + i * c;
  const float hn = tanhf(cn) * o;
  Cnew[idx] = cn;
  Hnew[idx] = hn;
  Abf[(size_t)b * DIM_K + DIM_INPUT + j] = f2bf(hn);
}

// ---------------------------------------------------------------------------
// In-place row-wise log_softmax: one 256-thread workgroup per row of 50257.
// ---------------------------------------------------------------------------
__global__ __launch_bounds__(256)
void logsoftmax_rows(float* __restrict__ logits) {
  float* x = logits + (size_t)blockIdx.x * VOCAB;
  __shared__ float red[256];
  const int tid = threadIdx.x;

  float m = -INFINITY;
  for (int i = tid; i < VOCAB; i += 256) m = fmaxf(m, x[i]);
  red[tid] = m; __syncthreads();
  for (int s = 128; s > 0; s >>= 1) {
    if (tid < s) red[tid] = fmaxf(red[tid], red[tid + s]);
    __syncthreads();
  }
  m = red[0]; __syncthreads();

  float sum = 0.0f;
  for (int i = tid; i < VOCAB; i += 256) sum += __expf(x[i] - m);
  red[tid] = sum; __syncthreads();
  for (int s = 128; s > 0; s >>= 1) {
    if (tid < s) red[tid] += red[tid + s];
    __syncthreads();
  }
  const float lse = m + __logf(red[0]);
  __syncthreads();

  for (int i = tid; i < VOCAB; i += 256) x[i] -= lse;
}

// ---------------------------------------------------------------------------
extern "C" void kernel_launch(void* const* d_in, const int* in_sizes, int n_in,
                              void* d_out, int out_size, void* d_ws, size_t ws_size,
                              hipStream_t stream) {
  const float* input = (const float*)d_in[0];
  const float* H     = (const float*)d_in[1];
  const float* C     = (const float*)d_in[2];
  const float* W_F   = (const float*)d_in[3];
  const float* b_F   = (const float*)d_in[4];
  const float* W_I   = (const float*)d_in[5];
  const float* b_I   = (const float*)d_in[6];
  const float* W_C   = (const float*)d_in[7];
  const float* b_C   = (const float*)d_in[8];
  const float* W_O   = (const float*)d_in[9];
  const float* b_O   = (const float*)d_in[10];
  const float* W_out = (const float*)d_in[11];
  const float* b_out = (const float*)d_in[12];

  float* outp   = (float*)d_out;
  float* logits = outp;                                      // [4096, 50257]
  float* Hnew   = outp + (size_t)BATCH * VOCAB;              // [4096, 1024]
  float* Cnew   = Hnew + (size_t)BATCH * DIM_HIDDEN;         // [4096, 1024]

  __bf16* Abf = (__bf16*)d_ws;                               // [4096, 2048] bf16 (16 MB)
  float*  G   = (float*)((char*)d_ws +
                         (size_t)BATCH * DIM_K * sizeof(__bf16)); // [4096, 4096] (64 MB)

  dim3 blk(256);

  // 0) pack concat(input, H) to bf16 (L2-resident A matrix)
  pack_a_kernel<<<(BATCH * DIM_HIDDEN) / 256, blk, 0, stream>>>(input, H, Abf);

  // 1) fused 4-gate GEMM + activation epilogue
  wmma_gemm_kernel<0><<<dim3(BATCH / 128, NGATE / 128), blk, 0, stream>>>(
      Abf, W_F, W_I, W_C, W_O, b_F, b_I, b_C, b_O, G, NGATE, NGATE);

  // 2) pointwise LSTM update (H_new/C_new into d_out, bf16 H_new into Abf)
  lstm_pointwise<<<(BATCH * DIM_HIDDEN) / 256, blk, 0, stream>>>(G, C, Hnew, Cnew, Abf);

  // 3) vocab projection GEMM (W_out streamed NT once; A from L2)
  wmma_gemm_kernel<1><<<dim3(BATCH / 128, (VOCAB + 127) / 128), blk, 0, stream>>>(
      Abf, W_out, nullptr, nullptr, nullptr,
      b_out, nullptr, nullptr, nullptr, logits, VOCAB, VOCAB);

  // 4) in-place log_softmax over each row
  logsoftmax_rows<<<BATCH, blk, 0, stream>>>(logits);
}
